// Linear4bit_66005057405786
// MI455X (gfx1250) — matrix operand order repaired and checked by reading, not compiled, not run on hardware
//
#include <hip/hip_runtime.h>

typedef __attribute__((ext_vector_type(16))) _Float16 v16h;
typedef __attribute__((ext_vector_type(8)))  float    v8f;
typedef __attribute__((ext_vector_type(4)))  float    v4f;
typedef __attribute__((ext_vector_type(4)))  int      v4i;

#define IN_F   4096
#define OUT_F  14336
#define NWAVES 8
#define KCHUNK (IN_F / NWAVES)   /* 512 */
#define KSTEPS (KCHUNK / 32)     /* 16  */

// NF4 codebook (bitsandbytes normal-float 4-bit quantiles)
__device__ const float NF4_TBL[16] = {
    -1.0f, -0.6961928009986877f, -0.5250730514526367f, -0.39491748809814453f,
    -0.28444138169288635f, -0.18477343022823334f, -0.09105003625154495f, 0.0f,
    0.07958029955625534f, 0.16093020141124725f, 0.24611230194568634f,
    0.33791524171829224f, 0.44070982933044434f, 0.5626170039176941f,
    0.7229568362236023f, 1.0f};

// Dequant two int codes -> packed f16 pair, via lane-crossbar codebook gather.
#define DQ2(dst, i, cA, cB, scl)                                                   \
  {                                                                                \
    float fa = __int_as_float(                                                     \
                   __builtin_amdgcn_ds_bpermute((cA) << 2, cb_bits)) * (scl);      \
    float fb = __int_as_float(                                                     \
                   __builtin_amdgcn_ds_bpermute((cB) << 2, cb_bits)) * (scl);      \
    auto p = __builtin_amdgcn_cvt_pkrtz(fa, fb);                                   \
    dst[(i)] = (_Float16)p[0];                                                     \
    dst[(i) + 1] = (_Float16)p[1];                                                 \
  }

// Build one 16xK=32 B fragment from a row of int32 codes.
#define BFRAG(dst, rowptr, scl)                                                    \
  {                                                                                \
    const v4i* cp = (const v4i*)((rowptr) + k0 + kg * 16);                         \
    v4i c0 = cp[0], c1 = cp[1], c2 = cp[2], c3 = cp[3];                            \
    DQ2(dst, 0,  c0.x, c0.y, scl); DQ2(dst, 2,  c0.z, c0.w, scl);                  \
    DQ2(dst, 4,  c1.x, c1.y, scl); DQ2(dst, 6,  c1.z, c1.w, scl);                  \
    DQ2(dst, 8,  c2.x, c2.y, scl); DQ2(dst, 10, c2.z, c2.w, scl);                  \
    DQ2(dst, 12, c3.x, c3.y, scl); DQ2(dst, 14, c3.z, c3.w, scl);                  \
  }

// Convert 4 consecutive f32 -> 4 consecutive f16 fragment elements.
#define CVT4(dst, i, v)                                                            \
  {                                                                                \
    auto p0 = __builtin_amdgcn_cvt_pkrtz((v).x, (v).y);                            \
    auto p1 = __builtin_amdgcn_cvt_pkrtz((v).z, (v).w);                            \
    dst[(i)] = (_Float16)p0[0]; dst[(i) + 1] = (_Float16)p0[1];                    \
    dst[(i) + 2] = (_Float16)p1[0]; dst[(i) + 3] = (_Float16)p1[1];                \
  }

// Build one 16(M)xK=32 A fragment from x rows (contiguous K order, same as B):
// elems 0..15 <- K = k0 + kg*16 + [0..15]
#define AFRAG(dst, xrow)                                                           \
  {                                                                                \
    v4f t0 = *(const v4f*)((xrow) + k0 + kg * 16);                                 \
    v4f t1 = *(const v4f*)((xrow) + k0 + kg * 16 + 4);                             \
    v4f t2 = *(const v4f*)((xrow) + k0 + kg * 16 + 8);                             \
    v4f t3 = *(const v4f*)((xrow) + k0 + kg * 16 + 12);                            \
    CVT4(dst, 0, t0); CVT4(dst, 4, t1); CVT4(dst, 8, t2); CVT4(dst, 12, t3);       \
  }

__global__ __launch_bounds__(256) void nf4_linear_kernel(
    const float* __restrict__ x, const int* __restrict__ codes,
    const float* __restrict__ absmax, const float* __restrict__ bias,
    float* __restrict__ out) {
  // 8 waves x 4 tiles x (8 VGPR x 32 lane) partials = 32 KB
  __shared__ float red[NWAVES * 4 * 8 * 32];

  const int tid  = threadIdx.x;
  const int lane = tid & 31;
  const int wave = tid >> 5;
  const int nl   = lane & 15;  // N (or M) index within tile
  const int kg   = lane >> 4;  // K-group select (0/1)
  const int O    = blockIdx.x * 32;  // output-feature tile base (2 N-tiles)

  // Materialize codebook into lanes 0..15 (one rodata load, kernel lifetime).
  const int cb_bits = __float_as_int(NF4_TBL[nl]);

  const int k_base = wave * KCHUNK;

  const int*   crow0 = codes  + (size_t)(O + nl) * IN_F;
  const int*   crow1 = codes  + (size_t)(O + 16 + nl) * IN_F;
  const float* am0   = absmax + (size_t)(O + nl) * (IN_F / 64);
  const float* am1   = absmax + (size_t)(O + 16 + nl) * (IN_F / 64);
  const float* x0    = x + (size_t)nl * IN_F;          // M-tile 0 rows 0..15
  const float* x1    = x + (size_t)(nl + 16) * IN_F;   // M-tile 1 rows 16..31

  v8f acc00 = {};  // (M-tile 0, N-tile 0)
  v8f acc01 = {};  // (M-tile 0, N-tile 1)
  v8f acc10 = {};  // (M-tile 1, N-tile 0)
  v8f acc11 = {};  // (M-tile 1, N-tile 1)

#pragma unroll 1
  for (int s = 0; s < KSTEPS; ++s) {
    const int k0 = k_base + s * 32;

    // Prefetch next step's code streams (global_prefetch_b8).
    if (s + 1 < KSTEPS) {
      __builtin_prefetch(crow0 + k0 + 32 + kg * 16, 0, 1);
      __builtin_prefetch(crow1 + k0 + 32 + kg * 16, 0, 1);
    }

    // One absmax per lane per 32-wide step (32-aligned run fits one 64-block).
    const float scale0 = am0[k0 >> 6];
    const float scale1 = am1[k0 >> 6];

    // ---- Two B fragments: K = k0 + kg*16 + [0..15], N = nl (+16) ----
    v16h bfrag0, bfrag1;
    BFRAG(bfrag0, crow0, scale0);
    BFRAG(bfrag1, crow1, scale1);

    // ---- Two A fragments (M rows 0..15 and 16..31) ----
    v16h a0, a1;
    AFRAG(a0, x0);
    AFRAG(a1, x1);

    // 2x2 tile: every fragment feeds two WMMAs.
    acc00 = __builtin_amdgcn_wmma_f32_16x16x32_f16(false, a0, false, bfrag0,
                                                   (short)0, acc00, false, false);
    acc10 = __builtin_amdgcn_wmma_f32_16x16x32_f16(false, a1, false, bfrag0,
                                                   (short)0, acc10, false, false);
    acc01 = __builtin_amdgcn_wmma_f32_16x16x32_f16(false, a0, false, bfrag1,
                                                   (short)0, acc01, false, false);
    acc11 = __builtin_amdgcn_wmma_f32_16x16x32_f16(false, a1, false, bfrag1,
                                                   (short)0, acc11, false, false);
  }

  // ---- Split-K reduction across the 8 waves via LDS ----
  // Tile index t = tm*2 + tn.
#pragma unroll
  for (int r = 0; r < 8; ++r) {
    red[((wave * 4 + 0) * 8 + r) * 32 + lane] = acc00[r];
    red[((wave * 4 + 1) * 8 + r) * 32 + lane] = acc01[r];
    red[((wave * 4 + 2) * 8 + r) * 32 + lane] = acc10[r];
    red[((wave * 4 + 3) * 8 + r) * 32 + lane] = acc11[r];
  }
  __syncthreads();

  if (wave < 4) {
    const int tm = wave >> 1;
    const int tn = wave & 1;
    const int o  = O + tn * 16 + nl;
    const float vb = bias[o];
#pragma unroll
    for (int r = 0; r < 8; ++r) {
      float sum = vb;
#pragma unroll
      for (int w = 0; w < NWAVES; ++w)
        sum += red[((w * 4 + wave) * 8 + r) * 32 + lane];
      // C/D layout: VGPR r -> M = r (lanes 0-15), M = 8+r (lanes 16-31)
      const int m = tm * 16 + r + kg * 8;
      out[(size_t)m * OUT_F + o] = sum;
    }
  }
}

extern "C" void kernel_launch(void* const* d_in, const int* in_sizes, int n_in,
                              void* d_out, int out_size, void* d_ws,
                              size_t ws_size, hipStream_t stream) {
  (void)in_sizes; (void)n_in; (void)d_ws; (void)ws_size; (void)out_size;
  const float* x      = (const float*)d_in[0];
  const int*   codes  = (const int*)d_in[1];
  const float* absmax = (const float*)d_in[2];
  const float* bias   = (const float*)d_in[3];
  float*       out    = (float*)d_out;

  dim3 grid(OUT_F / 32);  // 448 blocks, each covering 32 output features
  dim3 block(NWAVES * 32);
  nf4_linear_kernel<<<grid, block, 0, stream>>>(x, codes, absmax, bias, out);
}
